// MultiHeadAttention_28887950033485
// MI455X (gfx1250) — matrix operand order repaired and checked by reading, not compiled, not run on hardware
//
#include <hip/hip_runtime.h>
#include <hip/hip_bf16.h>

// ---------------------------------------------------------------------------
// MHA forward for MI455X (gfx1250, wave32, WMMA + async-to-LDS).
// Pipeline: [qkv_proj bf16-WMMA GEMM] -> [flash attention bf16-WMMA, K/V tiles
//           staged with GLOBAL_LOAD_ASYNC_TO_LDS_B32] -> [out proj + bias].
// Workspace (bf16/ushort): [0) Q [b,h,n,64]  [SZ) K [b,h,n,64]
//                          [2*SZ) V TRANSPOSED [b,h,64,n]  [3*SZ) O [b,n,1024]
// SZ = 4096*1024 elements; total ws use = 32 MB.
// ---------------------------------------------------------------------------

#define N_HEADC 16
#define D_HEADC 64
#define D_MODELC 1024
#define BATCHC 2
#define SEQC 2048
#define ROWSC (BATCHC * SEQC)                       // 4096
#define TENSOR_ELEMS ((size_t)ROWSC * D_MODELC)     // 4 Mi elements

typedef __bf16 bhalf;
typedef __attribute__((ext_vector_type(16))) bhalf v16bf;
typedef __attribute__((ext_vector_type(8)))  float v8f;

union Frag16 { v16bf v; unsigned int u[8]; };

__device__ __forceinline__ unsigned short f2bf(float f) {
  return __builtin_bit_cast(unsigned short, (bhalf)f);
}
__device__ __forceinline__ unsigned int pack2bf(float lo, float hi) {
  return (unsigned int)f2bf(lo) | ((unsigned int)f2bf(hi) << 16);
}

__device__ __forceinline__ v8f wmma_bf16(const Frag16& a, const Frag16& b, v8f c) {
  return __builtin_amdgcn_wmma_f32_16x16x32_bf16(
      /*neg_a=*/false, a.v, /*neg_b=*/false, b.v,
      /*c_mod=*/(short)0, c, /*reuse_a=*/false, /*reuse_b=*/false);
}

// ushort index of A-matrix element (m=0..15, k=0..31) within a 512-ushort
// per-wave region laid out as [8 vgpr][32 lane] dwords (ISA 16-bit A 16x32).
// k and k^1 share one dword (bit0 = k&1).
__device__ __forceinline__ int a_slot(int m, int k) {
  int v  = ((k >> 4) & 1) * 4 + ((k & 7) >> 1);
  int ln = m + (((k >> 3) & 1) << 4);
  return ((v * 32 + ln) << 1) | (k & 1);
}

// ushort index of B-matrix element (k=0..31, n=0..63) within a region laid
// out as [n/16 tiles][8 vgpr][32 lane] dwords (ISA 16-bit B 32x16 per tile).
// k and k^1 share one dword (bit0 = k&1).
__device__ __forceinline__ int b_slot(int k, int n) {
  int t  = n >> 4;
  int ln = (n & 15) | (((k >> 3) & 1) << 4);
  int v  = ((k >> 4) & 1) * 4 + ((k & 7) >> 1);
  return (((t * 8 + v) * 32 + ln) << 1) | (k & 1);
}

// Per-lane async DMA: LDS[lds_addr] = *(b32*)(base + goff).  ASYNCcnt-tracked.
__device__ __forceinline__ void async_ld_b32(unsigned int lds_addr,
                                             unsigned int goff,
                                             const unsigned short* base) {
  asm volatile("global_load_async_to_lds_b32 %0, %1, %2"
               :
               : "v"(lds_addr), "v"(goff), "s"(base)
               : "memory");
}
// Low 32 bits of a generic pointer to __shared__ == LDS byte address.
__device__ __forceinline__ unsigned int lds_addr_of(const void* p) {
  return (unsigned int)(unsigned long long)p;
}

// ---------------------------------------------------------------------------
// Kernel 1: Q/K/V projections.  grid = (64 Mtiles, 16 Ntiles, 3), block 128.
// C[64x64] = X[64x1024] * W[1024x64], bf16 WMMA, f32 accumulate.
// Q/K written [b,h,n,d]; V written transposed [b,h,d,n] (feeds async staging).
// ---------------------------------------------------------------------------
__global__ __launch_bounds__(128) void qkv_proj_kernel(
    const float* __restrict__ qry, const float* __restrict__ key,
    const float* __restrict__ val, const float* __restrict__ Wq,
    const float* __restrict__ Wk,  const float* __restrict__ Wv,
    unsigned short* __restrict__ qkv) {
  const int z = blockIdx.z;
  const float* X = (z == 0) ? qry : (z == 1) ? key : val;
  const float* W = (z == 0) ? Wq  : (z == 1) ? Wk  : Wv;
  unsigned short* Dst = qkv + (size_t)z * TENSOR_ELEMS;

  const int row0 = blockIdx.x * 64;
  const int nt0  = blockIdx.y * 64;
  const int tid  = threadIdx.x;
  const int wave = tid >> 5, lane = tid & 31;
  const int hx = lane >> 4, l15 = lane & 15;

  __shared__ unsigned int ldsA[4 * 256];   // 4 waves  x (8 vgpr x 32 lane)
  __shared__ unsigned int ldsB[4 * 256];   // 4 ntiles x (8 vgpr x 32 lane)

  v8f acc[4];
#pragma unroll
  for (int t = 0; t < 4; t++)
#pragma unroll
    for (int r = 0; r < 8; r++) acc[t][r] = 0.0f;

  for (int kk = 0; kk < D_MODELC; kk += 32) {
    __syncthreads();
    {  // stage X tile 64x32 (f32 -> bf16), packed-pair dword stores
      int r  = tid >> 1;
      int c0 = (tid & 1) << 4;
      const float* src = X + (size_t)(row0 + r) * D_MODELC + kk + c0;
      float f[16];
#pragma unroll
      for (int j = 0; j < 16; j++) f[j] = src[j];
      unsigned int* dst = &ldsA[(r >> 4) * 256];
      int m = r & 15;
#pragma unroll
      for (int j = 0; j < 8; j++)
        dst[a_slot(m, c0 + 2 * j) >> 1] = pack2bf(f[2 * j], f[2 * j + 1]);
    }
    {  // stage W tile 32x64 (f32 -> bf16): each thread does a 2k x 8n patch
      int kp = tid >> 3;              // k pair 0..15 -> rows 2kp, 2kp+1
      int n0 = (tid & 7) << 3;        // n offset
      const float* s0 = W + (size_t)(kk + 2 * kp) * D_MODELC + nt0 + n0;
      const float* s1 = s0 + D_MODELC;
      float f0[8], f1[8];
#pragma unroll
      for (int j = 0; j < 8; j++) { f0[j] = s0[j]; f1[j] = s1[j]; }
#pragma unroll
      for (int j = 0; j < 8; j++)
        ldsB[b_slot(2 * kp, n0 + j) >> 1] = pack2bf(f0[j], f1[j]);
    }
    __syncthreads();

    Frag16 a;
#pragma unroll
    for (int v = 0; v < 8; v++) a.u[v] = ldsA[wave * 256 + v * 32 + lane];
#pragma unroll
    for (int t = 0; t < 4; t++) {
      Frag16 bf;
#pragma unroll
      for (int v = 0; v < 8; v++) bf.u[v] = ldsB[(t * 8 + v) * 32 + lane];
      acc[t] = wmma_bf16(a, bf, acc[t]);
    }
  }

  // Epilogue: head-split bf16 store.  V (z==2) goes out transposed [b,h,d,n].
#pragma unroll
  for (int t = 0; t < 4; t++) {
    int col = nt0 + t * 16 + l15;
    int h = col >> 6, d = col & 63;
#pragma unroll
    for (int r = 0; r < 8; r++) {
      int gi = row0 + wave * 16 + r + 8 * hx;
      int bb = gi >> 11, i = gi & 2047;
      size_t idx;
      if (z == 2)
        idx = ((size_t)(bb * N_HEADC + h) * D_HEADC + d) * SEQC + i;
      else
        idx = ((size_t)(bb * N_HEADC + h) * SEQC + i) * D_HEADC + d;
      Dst[idx] = f2bf(acc[t][r]);
    }
  }
}

// ---------------------------------------------------------------------------
// Kernel 2: flash attention.  grid = (SEQ/64, b*h), block 128 (4 waves).
// Each wave owns 16 query rows.  K/V 32-key tiles are DMA'd straight into
// WMMA-fragment-ordered LDS with global_load_async_to_lds_b32 (ASYNCcnt).
// ---------------------------------------------------------------------------
__global__ __launch_bounds__(128) void flash_attn_kernel(
    const unsigned short* __restrict__ qkv,
    const unsigned char* __restrict__ mask,   // jax bool: 1 byte per key
    unsigned short* __restrict__ obf) {
  const int bh = blockIdx.y;
  const int bb = bh >> 4, h = bh & 15;
  const int row0 = blockIdx.x * 64;
  const int tid = threadIdx.x;
  const int wave = tid >> 5, lane = tid & 31;
  const int hx = lane >> 4, l15 = lane & 15;

  const unsigned short* Q  = qkv + (size_t)bh * SEQC * D_HEADC;
  const unsigned short* Kh = qkv + TENSOR_ELEMS + (size_t)bh * SEQC * D_HEADC;
  const unsigned short* Vt = qkv + 2 * TENSOR_ELEMS + (size_t)bh * D_HEADC * SEQC;
  const unsigned char* mk = mask + bb * SEQC;

  __shared__ unsigned int ldsK[1024];  // [2 kchunk][2 jtile][8 vgpr][32 lane]
  __shared__ unsigned int ldsV[1024];  // [4 dtile][8 vgpr][32 lane]
  __shared__ unsigned int ldsP[1024];  // 4 waves x A-fragment of P (16x32)

  // Q A-fragments (16 rows x 64, two 32-wide K chunks), loaded once.
  Frag16 qf[2];
  {
    const unsigned short* qrow = Q + (size_t)(row0 + wave * 16 + l15) * D_HEADC;
#pragma unroll
    for (int c = 0; c < 2; c++)
#pragma unroll
      for (int v = 0; v < 8; v++) {
        int kb = c * 32 + ((v >> 2) << 4) + (hx << 3) + ((v & 3) << 1);
        qf[c].u[v] = *(const unsigned int*)(qrow + kb);
      }
  }

  // Per-lane async-staging coordinates (loop-invariant).
  const int krr = tid >> 2;            // K: key row 0..31
  const int kc0 = (tid & 3) << 4;      // K: d base
  const int vdd = tid >> 1;            // V: d row 0..63
  const int vk0 = (tid & 1) << 4;      // V: key base

  v8f oacc[4];
#pragma unroll
  for (int t = 0; t < 4; t++)
#pragma unroll
    for (int r = 0; r < 8; r++) oacc[t][r] = 0.0f;
  float mrun[8], lrun[8];
#pragma unroll
  for (int r = 0; r < 8; r++) { mrun[r] = -1e30f; lrun[r] = 0.0f; }

  for (int jb = 0; jb < SEQC / 32; jb++) {
    __syncthreads();
    // --- async DMA: K tile (memory dword = (d,d+1) pair = fragment dword) ---
#pragma unroll
    for (int q = 0; q < 8; q++) {
      int d = kc0 + 2 * q;
      unsigned int goff = (unsigned int)(((jb * 32 + krr) * D_HEADC + d) * 2);
      unsigned int la = lds_addr_of(&ldsK[(d >> 5) * 512 + (b_slot(d & 31, krr) >> 1)]);
      async_ld_b32(la, goff, Kh);
    }
    // --- async DMA: V tile (V transposed: dword = (kp,kp+1) pair) ---
#pragma unroll
    for (int q = 0; q < 8; q++) {
      int kp = vk0 + 2 * q;
      unsigned int goff = (unsigned int)((vdd * SEQC + jb * 32 + kp) * 2);
      unsigned int la = lds_addr_of(&ldsV[b_slot(kp, vdd) >> 1]);
      async_ld_b32(la, goff, Vt);
    }
    asm volatile("s_wait_asynccnt 0x0" ::: "memory");
    __syncthreads();

    // S = Q * K^T  (two 16x16 j-tiles, accumulated over two 32-wide d-chunks)
    v8f s0, s1;
#pragma unroll
    for (int r = 0; r < 8; r++) { s0[r] = 0.0f; s1[r] = 0.0f; }
#pragma unroll
    for (int c = 0; c < 2; c++) {
      Frag16 kb0, kb1;
#pragma unroll
      for (int v = 0; v < 8; v++) kb0.u[v] = ldsK[((c * 2 + 0) * 8 + v) * 32 + lane];
#pragma unroll
      for (int v = 0; v < 8; v++) kb1.u[v] = ldsK[((c * 2 + 1) * 8 + v) * 32 + lane];
      s0 = wmma_bf16(qf[c], kb0, s0);
      s1 = wmma_bf16(qf[c], kb1, s1);
    }

    // scale, key-padding mask, online softmax (rows 0-7 live in lanes 0-15,
    // rows 8-15 in lanes 16-31; shfl_xor masks <=8 stay within each half).
    const float NEG = -1e30f;
    bool ok0 = mk[jb * 32 + l15] != 0;
    bool ok1 = mk[jb * 32 + 16 + l15] != 0;
    float alpha[8];
#pragma unroll
    for (int r = 0; r < 8; r++) {
      float a0 = ok0 ? s0[r] * 0.125f : NEG;
      float a1 = ok1 ? s1[r] * 0.125f : NEG;
      float mx = fmaxf(a0, a1);
#pragma unroll
      for (int off = 8; off > 0; off >>= 1) mx = fmaxf(mx, __shfl_xor(mx, off, 32));
      float mnew = fmaxf(mrun[r], mx);
      float al = __expf(mrun[r] - mnew);
      mrun[r] = mnew;
      float p0 = __expf(a0 - mnew), p1 = __expf(a1 - mnew);
      s0[r] = p0; s1[r] = p1;
      float ls = p0 + p1;
#pragma unroll
      for (int off = 8; off > 0; off >>= 1) ls += __shfl_xor(ls, off, 32);
      lrun[r] = lrun[r] * al + ls;
      alpha[r] = al;
    }
#pragma unroll
    for (int t = 0; t < 4; t++)
#pragma unroll
      for (int r = 0; r < 8; r++) oacc[t][r] *= alpha[r];

    {  // transpose P from C/D layout to A layout via per-wave LDS region
      unsigned short* dP = (unsigned short*)&ldsP[wave * 256];
#pragma unroll
      for (int r = 0; r < 8; r++) {
        int m = r + 8 * hx;
        dP[a_slot(m, l15)]      = f2bf(s0[r]);
        dP[a_slot(m, 16 + l15)] = f2bf(s1[r]);
      }
    }
    asm volatile("s_wait_dscnt 0x0" ::: "memory");

    Frag16 pf;
#pragma unroll
    for (int v = 0; v < 8; v++) pf.u[v] = ldsP[wave * 256 + v * 32 + lane];
#pragma unroll
    for (int t = 0; t < 4; t++) {
      Frag16 vf;
#pragma unroll
      for (int v = 0; v < 8; v++) vf.u[v] = ldsV[(t * 8 + v) * 32 + lane];
      oacc[t] = wmma_bf16(pf, vf, oacc[t]);
    }
  }

  // O /= l, write bf16 in [b, n, h*64+d] layout for the output projection.
#pragma unroll
  for (int r = 0; r < 8; r++) {
    float inv = 1.0f / lrun[r];
    int i = row0 + wave * 16 + r + 8 * hx;
    size_t base = ((size_t)(bb * SEQC + i)) * (N_HEADC * D_HEADC) + h * D_HEADC;
#pragma unroll
    for (int t = 0; t < 4; t++)
      obf[base + t * 16 + l15] = f2bf(oacc[t][r] * inv);
  }
}

// ---------------------------------------------------------------------------
// Kernel 3: output projection + bias.  grid = (64, 16), block 128.
// out[64x64 f32] = O_bf16[64x1024] * Wo[1024x64] + bo
// ---------------------------------------------------------------------------
__global__ __launch_bounds__(128) void out_proj_kernel(
    const unsigned short* __restrict__ obf,
    const float* __restrict__ Wo, const float* __restrict__ bo,
    float* __restrict__ out) {
  const int row0 = blockIdx.x * 64;
  const int nt0  = blockIdx.y * 64;
  const int tid  = threadIdx.x;
  const int wave = tid >> 5, lane = tid & 31;
  const int hx = lane >> 4, l15 = lane & 15;

  __shared__ unsigned int ldsA[4 * 256];
  __shared__ unsigned int ldsB[4 * 256];

  v8f acc[4];
#pragma unroll
  for (int t = 0; t < 4; t++)
#pragma unroll
    for (int r = 0; r < 8; r++) acc[t][r] = 0.0f;

  for (int kk = 0; kk < D_MODELC; kk += 32) {
    __syncthreads();
    {  // stage O tile 64x32 (bf16): memory dword == fragment dword, pure copy
      int r  = tid >> 1;
      int c0 = (tid & 1) << 4;
      const unsigned int* src =
          (const unsigned int*)(obf + (size_t)(row0 + r) * D_MODELC + kk + c0);
      unsigned int u[8];
#pragma unroll
      for (int j = 0; j < 8; j++) u[j] = src[j];
      unsigned int* dst = &ldsA[(r >> 4) * 256];
      int m = r & 15;
#pragma unroll
      for (int j = 0; j < 8; j++) dst[a_slot(m, c0 + 2 * j) >> 1] = u[j];
    }
    {  // stage Wo tile 32x64 (f32 -> bf16): 2k x 8n patch per thread
      int kp = tid >> 3;
      int n0 = (tid & 7) << 3;
      const float* s0 = Wo + (size_t)(kk + 2 * kp) * D_MODELC + nt0 + n0;
      const float* s1 = s0 + D_MODELC;
      float f0[8], f1[8];
#pragma unroll
      for (int j = 0; j < 8; j++) { f0[j] = s0[j]; f1[j] = s1[j]; }
#pragma unroll
      for (int j = 0; j < 8; j++)
        ldsB[b_slot(2 * kp, n0 + j) >> 1] = pack2bf(f0[j], f1[j]);
    }
    __syncthreads();

    Frag16 a;
#pragma unroll
    for (int v = 0; v < 8; v++) a.u[v] = ldsA[wave * 256 + v * 32 + lane];
#pragma unroll
    for (int t = 0; t < 4; t++) {
      Frag16 bf;
#pragma unroll
      for (int v = 0; v < 8; v++) bf.u[v] = ldsB[(t * 8 + v) * 32 + lane];
      acc[t] = wmma_bf16(a, bf, acc[t]);
    }
  }

#pragma unroll
  for (int t = 0; t < 4; t++) {
    int col = nt0 + t * 16 + l15;
    float bias = bo[col];
#pragma unroll
    for (int r = 0; r < 8; r++) {
      int gi = row0 + wave * 16 + r + 8 * hx;
      out[(size_t)gi * D_MODELC + col] = acc[t][r] + bias;
    }
  }
}

// ---------------------------------------------------------------------------
extern "C" void kernel_launch(void* const* d_in, const int* in_sizes, int n_in,
                              void* d_out, int out_size, void* d_ws, size_t ws_size,
                              hipStream_t stream) {
  const float* qry  = (const float*)d_in[0];
  const float* key  = (const float*)d_in[1];
  const float* val  = (const float*)d_in[2];
  const unsigned char* mask = (const unsigned char*)d_in[3];  // jax bool = 1B
  const float* Wq = (const float*)d_in[4];
  const float* Wk = (const float*)d_in[5];
  const float* Wv = (const float*)d_in[6];
  const float* Wo = (const float*)d_in[7];
  const float* bo = (const float*)d_in[8];

  unsigned short* ws16 = (unsigned short*)d_ws;      // needs 32 MB
  unsigned short* qkv_bf = ws16;                     // 3 * TENSOR_ELEMS
  unsigned short* o_bf   = ws16 + 3 * TENSOR_ELEMS;  // 1 * TENSOR_ELEMS

  qkv_proj_kernel<<<dim3(ROWSC / 64, D_MODELC / 64, 3), 128, 0, stream>>>(
      qry, key, val, Wq, Wk, Wv, qkv_bf);
  flash_attn_kernel<<<dim3(SEQC / 64, BATCHC * N_HEADC), 128, 0, stream>>>(
      qkv_bf, mask, o_bf);
  out_proj_kernel<<<dim3(ROWSC / 64, D_MODELC / 64), 128, 0, stream>>>(
      o_bf, Wo, bo, (float*)d_out);
}
